// TDNNKalmanNet_13924283974264
// MI455X (gfx1250) — matrix operand order repaired and checked by reading, compiled, not run on hardware
//
#include <hip/hip_runtime.h>
#include <hip/hip_bf16.h>
#include <math.h>

// MI455X / gfx1250: wave32, WMMA bf16 16x16x32 (f32 accum), TDM staging,
// wave-parallel layernorm/head reductions via shuffles.

typedef __attribute__((ext_vector_type(16))) __bf16 v16bf;
typedef __attribute__((ext_vector_type(8)))  __bf16 v8bf;
typedef __attribute__((ext_vector_type(8)))  float  v8f;
typedef __attribute__((ext_vector_type(4)))  unsigned int u32x4;
typedef __attribute__((ext_vector_type(8)))  int i32x8;
typedef __attribute__((ext_vector_type(4)))  int i32x4;

#define NQ    64
#define ND    64
#define NH    192
#define NWOUT 32
#define PI_F  3.14159265358979323846f

// ---- bf16 weight arena (element offsets inside workspace) ----
#define OFF_CIW   0                         // 192*192   conv_in  W[o][i]
#define OFF_WTAP  36864                     // 4 convs * 3 taps * 192*192, W[o][i] per tap
#define OFF_FGW   (36864 + 442368)          // 64*192
#define OFF_RIW   (OFF_FGW + 12288)         // 192*256
#define OFF_WIH   (OFF_RIW + 49152)         // 576*192
#define OFF_WHH   (OFF_WIH + 110592)        // 576*192
#define WB_TOTAL  (OFF_WHH + 110592)        // 761856 elems (~1.5 MB)

#if defined(__has_builtin)
#if __has_builtin(__builtin_amdgcn_tensor_load_to_lds) && __has_builtin(__builtin_amdgcn_s_wait_tensorcnt)
#define HAVE_TDM 1
#endif
#endif
#ifndef HAVE_TDM
#define HAVE_TDM 0
#endif

__device__ __forceinline__ float gelu_f(float x) {
    return 0.5f * x * (1.0f + erff(x * 0.70710678118654752440f));
}
__device__ __forceinline__ float sigm_f(float x) {
    return 1.0f / (1.0f + expf(-x));
}
// wave32 butterfly all-reduce (sum to every lane)
__device__ __forceinline__ float wsum(float v) {
#pragma unroll
    for (int o = 16; o > 0; o >>= 1) v += __shfl_xor(v, o, 32);
    return v;
}

// ---------------- WMMA fragment helpers (ISA 7.12.2 layouts) ----------------
__device__ __forceinline__ v16bf frag_a(const __bf16* rowp, int kbase, int hi) {
    const v8bf* p = (const v8bf*)(rowp + kbase + hi * 8);
    return __builtin_shufflevector(p[0], p[2], 0,1,2,3,4,5,6,7,8,9,10,11,12,13,14,15);
}
__device__ __forceinline__ v16bf frag_b(const __bf16* W, int ldk, int n, int kbase, int hi) {
    const v8bf* p = (const v8bf*)(W + (size_t)n * ldk + kbase + hi * 8);
    return __builtin_shufflevector(p[0], p[2], 0,1,2,3,4,5,6,7,8,9,10,11,12,13,14,15);
}
__device__ __forceinline__ v8f wmma_bf(v16bf a, v16bf b, v8f c) {
    return __builtin_amdgcn_wmma_f32_16x16x32_bf16(false, a, false, b, (short)0, c, false, false);
}

#if HAVE_TDM
// 1-D contiguous global->LDS copy through the Tensor Data Mover (ISA ch.8 D#).
__device__ __forceinline__ void tdm_load_1d(unsigned lds_byte_off, const void* gsrc, unsigned nbytes)
{
    unsigned long long ga = (unsigned long long)(size_t)gsrc;
    unsigned n8 = nbytes >> 3;                                // 8-byte elements
    u32x4 g0 = {};
    g0[0] = 1u;                                               // count=1 (valid user D#)
    g0[1] = lds_byte_off;                                     // lds_addr (bytes)
    g0[2] = (unsigned)(ga & 0xffffffffu);                     // global_addr[31:0]
    g0[3] = (unsigned)((ga >> 32) & 0x01ffffffu) | (2u << 30);// addr[56:32] | type=2
    i32x8 g1 = {};
    g1[0] = (int)(3u << 16);                                  // data_size = 8 bytes
    g1[1] = (int)((n8 & 0xffffu) << 16);                      // tensor_dim0[15:0]
    g1[2] = (int)(((n8 >> 16) & 0xffffu) | (1u << 16));       // tensor_dim0[31:16] | tensor_dim1=1
    g1[3] = (int)((n8 & 0xffffu) << 16);                      // tile_dim0 = n8 (fits 16b)
    g1[5] = (int)n8;                                          // tensor_dim0_stride
    i32x4 z4 = {};
#if defined(__clang_major__) && __clang_major__ >= 23
    i32x8 z8 = {};
    __builtin_amdgcn_tensor_load_to_lds(g0, g1, z4, z4, z8, 0);
#else
    __builtin_amdgcn_tensor_load_to_lds(g0, g1, z4, z4, 0);
#endif
}
#endif

// ---------------- params ----------------
struct P0 {
    const float *ciw, *b0c1w, *b0c2w, *b1c1w, *b1c2w, *fgw, *riw, *wih, *whh;
    __bf16* wb;
};
struct P1 {
    const float* x;
    const float *cib;
    const float *b0c1b, *b0c2b, *b0g, *b0b;
    const float *b1c1b, *b1c2b, *b1g, *b1b;
    const float *olg, *olb, *fgb, *frw, *frb;
    const __bf16* wb;
    float *xpost_g, *hlast_g;
};
struct P2 {
    const float *rib, *bih, *bhh, *rlg, *rlb, *fqw, *fqb;
    const __bf16* wb;
    const float *xpost_g, *hlast_g;
    float* out;
};

// ------------- kernel 0: one-time f32 -> bf16 weight repack -------------
__global__ void __launch_bounds__(256, 1) k0_prep(P0 p)
{
    const int i = blockIdx.x * blockDim.x + threadIdx.x;
    if (i < 36864)  p.wb[OFF_CIW + i] = (__bf16)p.ciw[i];
    if (i < 442368) {
        int c = i / (3 * 36864), r = i % (3 * 36864);
        int t = r / 36864, e = r % 36864;
        int o = e / NH, k = e - o * NH;
        const float* w = (c == 0) ? p.b0c1w : (c == 1) ? p.b0c2w : (c == 2) ? p.b1c1w : p.b1c2w;
        p.wb[OFF_WTAP + i] = (__bf16)w[o * (NH * 3) + k * 3 + t];
    }
    if (i < 12288)  p.wb[OFF_FGW + i] = (__bf16)p.fgw[i];
    if (i < 49152)  p.wb[OFF_RIW + i] = (__bf16)p.riw[i];
    if (i < 110592) {
        p.wb[OFF_WIH + i] = (__bf16)p.wih[i];
        p.wb[OFF_WHH + i] = (__bf16)p.whh[i];
    }
}

// K=3 dilated conv as WMMA GEMM, 3 taps, per-tap weights staged to LDS via TDM.
__device__ __forceinline__ void conv192_k3(
    const __bf16* src, __bf16* dst, __bf16* wt,
    const __bf16* wtap, const float* bias, int dil, int tid)
{
    const int wave = tid >> 5, lane = tid & 31;
    const int m = lane & 15, hi = lane >> 4;
    v8f zero = {};
    v8f acc[6];
#pragma unroll
    for (int i = 0; i < 6; ++i) acc[i] = zero;

    for (int t = 0; t < 3; ++t) {
        __syncthreads();
        const __bf16* wsrc = wtap + (size_t)t * (NH * NH);
#if HAVE_TDM
        if (tid < 32) {     // wave 0 issues the tensor DMA and waits on TENSORcnt
            tdm_load_1d((unsigned)(size_t)(void*)wt, wsrc, NH * NH * 2);
            __builtin_amdgcn_s_wait_tensorcnt(0);
        }
#else
        {
            const uint4* s4 = (const uint4*)wsrc;
            uint4* d4 = (uint4*)wt;
            for (int e = tid; e < (NH * NH) / 8; e += 256) d4[e] = s4[e];
        }
#endif
        __syncthreads();
        const int shift = (t - 1) * dil;
        for (int i6 = 0; i6 < 6; ++i6) {
            int idx = wave * 6 + i6;
            int rt = idx / 12, ct = idx % 12;
            int q = rt * 16 + m + shift;
            q = q < 0 ? 0 : (q > NQ - 1 ? NQ - 1 : q);
            const __bf16* arow = src + q * NH;
            int n = ct * 16 + m;
            for (int ks = 0; ks < 6; ++ks)
                acc[i6] = wmma_bf(frag_a(arow, ks * 32, hi),
                                  frag_b(wt, NH, n, ks * 32, hi), acc[i6]);
        }
    }
    __syncthreads();   // all reads of src done before (possibly aliased) writes
    for (int i6 = 0; i6 < 6; ++i6) {
        int idx = wave * 6 + i6;
        int rt = idx / 12, ct = idx % 12;
        int col = ct * 16 + m;
        float bv = bias[col];
#pragma unroll
        for (int r = 0; r < 8; ++r) {
            int row = rt * 16 + hi * 8 + r;
            dst[row * NH + col] = (__bf16)gelu_f(acc[i6][r] + bv);
        }
    }
    __syncthreads();
}

__device__ __forceinline__ void resblock(
    __bf16* hb, __bf16* yb, __bf16* wt, const __bf16* wtaps,
    const float* c1b, const float* c2b,
    const float* g, const float* bb, int dil, int tid)
{
    conv192_k3(hb, yb, wt, wtaps,                  c1b, dil, tid);
    conv192_k3(yb, yb, wt, wtaps + 3ull * NH * NH, c2b, dil, tid);
    // residual + layernorm: one wave per row, lanes stride channels
    const int wave = tid >> 5, lane = tid & 31;
    for (int rr = 0; rr < 8; ++rr) {
        const int base = (wave + rr * 8) * NH;
        float s = 0.0f, s2 = 0.0f;
        for (int i = lane; i < NH; i += 32) {
            float v = (float)hb[base + i] + (float)yb[base + i];
            s += v; s2 += v * v;
        }
        s = wsum(s); s2 = wsum(s2);
        float mean = s * (1.0f / NH);
        float inv = rsqrtf(s2 * (1.0f / NH) - mean * mean + 1e-5f);
        for (int i = lane; i < NH; i += 32) {
            float v = (float)hb[base + i] + (float)yb[base + i];
            hb[base + i] = (__bf16)((v - mean) * inv * g[i] + bb[i]);
        }
    }
    __syncthreads();
}

// ------------- kernel 1: front-end + Kalman scan, 1 batch element / WG -------------
__global__ void __launch_bounds__(256, 1) k1_front(P1 p)
{
    __shared__ float  xs[NQ * ND];      // 16 KB
    __shared__ __bf16 hb[NQ * NH];      // 24 KB
    __shared__ __bf16 yb[NQ * NH];      // 24 KB
    __shared__ __bf16 wt[NH * NH];      // 72 KB per-tap weight slice (TDM target)
    __shared__ float  kg[NQ * ND];      // 16 KB
    __shared__ float  rho_s[NQ];
    __shared__ float  phi_s[NQ];

    const int tid  = threadIdx.x;
    const int wave = tid >> 5, lane = tid & 31;
    const int m = lane & 15, hi = lane >> 4;
    const int b = blockIdx.x;

    const float* xg = p.x + (size_t)b * NQ * ND;
#if HAVE_TDM
    if (tid < 32) {
        tdm_load_1d((unsigned)(size_t)(void*)xs, xg, NQ * ND * 4);
        __builtin_amdgcn_s_wait_tensorcnt(0);
    }
#else
    for (int e = tid; e < NQ * ND; e += 256) xs[e] = xg[e];
#endif
    __syncthreads();

    // feats = [x, dy, ddy] -> yb
    for (int e = tid; e < NQ * NH; e += 256) {
        int q = e / NH, k = e - q * NH;
        float v;
        if (k < ND) {
            v = xs[q * ND + k];
        } else if (k < 2 * ND) {
            int c = k - ND;
            v = (q >= 1) ? xs[q * ND + c] - xs[(q - 1) * ND + c] : 0.0f;
        } else {
            int c = k - 2 * ND;
            float dq  = (q >= 1) ? xs[q * ND + c] - xs[(q - 1) * ND + c] : 0.0f;
            float dqm = (q >= 2) ? xs[(q - 1) * ND + c] - xs[(q - 2) * ND + c] : 0.0f;
            v = (q >= 1) ? dq - dqm : 0.0f;
        }
        yb[e] = (__bf16)v;
    }
    __syncthreads();

    // conv_in (1x1): hb = feats @ ciw^T + cib   (48 WMMA tiles / 8 waves)
    {
        const __bf16* ciwb = p.wb + OFF_CIW;
        v8f zero = {};
        v8f acc[6];
#pragma unroll
        for (int i = 0; i < 6; ++i) acc[i] = zero;
        for (int i6 = 0; i6 < 6; ++i6) {
            int idx = wave * 6 + i6;
            int rt = idx / 12, ct = idx % 12;
            const __bf16* arow = yb + (rt * 16 + m) * NH;
            int n = ct * 16 + m;
            for (int ks = 0; ks < 6; ++ks)
                acc[i6] = wmma_bf(frag_a(arow, ks * 32, hi),
                                  frag_b(ciwb, NH, n, ks * 32, hi), acc[i6]);
        }
        for (int i6 = 0; i6 < 6; ++i6) {
            int idx = wave * 6 + i6;
            int rt = idx / 12, ct = idx % 12;
            int col = ct * 16 + m;
            float bv = p.cib[col];
#pragma unroll
            for (int r = 0; r < 8; ++r) {
                int row = rt * 16 + hi * 8 + r;
                hb[row * NH + col] = (__bf16)(acc[i6][r] + bv);
            }
        }
    }
    const __bf16* wtaps = p.wb + OFF_WTAP;
    resblock(hb, yb, wt, wtaps,                  p.b0c1b, p.b0c2b, p.b0g, p.b0b, 1, tid);
    resblock(hb, yb, wt, wtaps + 6ull * NH * NH, p.b1c1b, p.b1c2b, p.b1g, p.b1b, 2, tid);

    // out_ln (in place on hb) + rho/phi heads: one wave per row
    for (int rr = 0; rr < 8; ++rr) {
        const int row = wave + rr * 8;
        const int base = row * NH;
        float s = 0.0f, s2 = 0.0f;
        for (int i = lane; i < NH; i += 32) {
            float v = (float)hb[base + i];
            s += v; s2 += v * v;
        }
        s = wsum(s); s2 = wsum(s2);
        float mean = s * (1.0f / NH);
        float inv = rsqrtf(s2 * (1.0f / NH) - mean * mean + 1e-5f);
        float s0 = 0.0f, s1 = 0.0f;
        for (int i = lane; i < NH; i += 32) {
            float o = ((float)hb[base + i] - mean) * inv * p.olg[i] + p.olb[i];
            hb[base + i] = (__bf16)o;
            s0 += o * p.frw[i];
            s1 += o * p.frw[NH + i];
        }
        s0 = wsum(s0); s1 = wsum(s1);
        if (lane == 0) {
            rho_s[row] = sigm_f(s0 + p.frb[0]) * 1.25f;
            phi_s[row] = PI_F * tanhf(s1 + p.frb[1]);
        }
    }
    __syncthreads();

    // Kg = sigmoid(h_seq @ fgw^T + fgb)  [64x64], 16 tiles / 8 waves
    {
        const __bf16* fgwb = p.wb + OFF_FGW;
        v8f zero = {};
        v8f acc[2]; acc[0] = zero; acc[1] = zero;
        for (int i2 = 0; i2 < 2; ++i2) {
            int idx = wave * 2 + i2;
            int rt = idx / 4, ct = idx % 4;
            const __bf16* arow = hb + (rt * 16 + m) * NH;
            int n = ct * 16 + m;
            for (int ks = 0; ks < 6; ++ks)
                acc[i2] = wmma_bf(frag_a(arow, ks * 32, hi),
                                  frag_b(fgwb, NH, n, ks * 32, hi), acc[i2]);
        }
        for (int i2 = 0; i2 < 2; ++i2) {
            int idx = wave * 2 + i2;
            int rt = idx / 4, ct = idx % 4;
            int col = ct * 16 + m;
            float bv = p.fgb[col];
#pragma unroll
            for (int r = 0; r < 8; ++r) {
                int row = rt * 16 + hi * 8 + r;
                kg[row * ND + col] = sigm_f(acc[i2][r] + bv);
            }
        }
    }
    __syncthreads();

    // Kalman scan: 32 lanes each own a (re, im) pair; 64 sequential steps.
    if (tid < 32) {
        float re = xs[tid], im = xs[32 + tid];
        for (int q = 0; q < NQ; ++q) {
            float r = rho_s[q], ph = phi_s[q];
            float cc = cosf(ph), ss = sinf(ph);
            float pre = r * (cc * re - ss * im);
            float pim = r * (ss * re + cc * im);
            float yr = xs[q * ND + tid], yi = xs[q * ND + 32 + tid];
            re = pre + kg[q * ND + tid]      * (yr - pre);
            im = pim + kg[q * ND + 32 + tid] * (yi - pim);
        }
        p.xpost_g[(size_t)b * ND + tid]      = re;
        p.xpost_g[(size_t)b * ND + 32 + tid] = im;
    }
    for (int c = tid; c < NH; c += 256)
        p.hlast_g[(size_t)b * NH + c] = (float)hb[63 * NH + c];
}

// ------------- kernel 2: 32-step GRU rollout, 16 batch rows / WG -------------
__global__ void __launch_bounds__(256, 1) k2_roll(P2 p)
{
    __shared__ float  hrf[16 * NH];
    __shared__ __bf16 cat[16 * 256];
    __shared__ __bf16 zb[16 * NH];
    __shared__ float  gi[16 * 576];
    __shared__ float  gh[16 * 576];
    __shared__ float  tmpf[16 * NH];
    __shared__ float  cur[16 * ND];
    __shared__ float  rho2[16];
    __shared__ float  phi2[16];

    const int tid = threadIdx.x;
    const int wave = tid >> 5, lane = tid & 31;
    const int m = lane & 15, hi = lane >> 4;
    const int b0 = blockIdx.x * 16;
    const __bf16* riwb = p.wb + OFF_RIW;
    const __bf16* wihb = p.wb + OFF_WIH;
    const __bf16* whhb = p.wb + OFF_WHH;

    if (lane == 0) {            // warm caches for the streamed bf16 weights
        __builtin_prefetch(riwb, 0, 0);
        __builtin_prefetch(wihb, 0, 0);
        __builtin_prefetch(whhb, 0, 0);
    }
#if HAVE_TDM
    if (tid < 32) {
        tdm_load_1d((unsigned)(size_t)(void*)hrf, p.hlast_g + (size_t)b0 * NH, 16 * NH * 4);
        tdm_load_1d((unsigned)(size_t)(void*)cur, p.xpost_g + (size_t)b0 * ND, 16 * ND * 4);
        __builtin_amdgcn_s_wait_tensorcnt(0);
    }
#else
    for (int e = tid; e < 16 * NH; e += 256) hrf[e] = p.hlast_g[(size_t)b0 * NH + e];
    for (int e = tid; e < 16 * ND; e += 256) cur[e] = p.xpost_g[(size_t)b0 * ND + e];
#endif
    __syncthreads();

    for (int step = 0; step < NWOUT; ++step) {
        for (int e = tid; e < 16 * 256; e += 256) {
            int row = e >> 8, c = e & 255;
            float v = (c < NH) ? hrf[row * NH + c] : cur[row * ND + (c - NH)];
            cat[e] = (__bf16)v;
        }
        __syncthreads();
        // Z = tanh(cat @ riw^T + rib)   [16x192], K=256
        for (int ct = wave; ct < 12; ct += 8) {
            v8f acc = {};
            const __bf16* arow = cat + m * 256;
            int n = ct * 16 + m;
            for (int ks = 0; ks < 8; ++ks)
                acc = wmma_bf(frag_a(arow, ks * 32, hi),
                              frag_b(riwb, 256, n, ks * 32, hi), acc);
            float bv = p.rib[n];
#pragma unroll
            for (int r = 0; r < 8; ++r)
                zb[(hi * 8 + r) * NH + n] = (__bf16)tanhf(acc[r] + bv);
        }
        __syncthreads();
        // GI = z @ wih^T + bih ; GH = h_r @ whh^T + bhh   [16x576], K=192
        for (int ct = wave; ct < 36; ct += 8) {
            v8f ai = {}, ah = {};
            const __bf16* zrow = zb + m * NH;
            const __bf16* hrow = cat + m * 256;     // cols 0..191 == h_r
            int n = ct * 16 + m;
            for (int ks = 0; ks < 6; ++ks) {
                ai = wmma_bf(frag_a(zrow, ks * 32, hi),
                             frag_b(wihb, NH, n, ks * 32, hi), ai);
                ah = wmma_bf(frag_a(hrow, ks * 32, hi),
                             frag_b(whhb, NH, n, ks * 32, hi), ah);
            }
            float b1 = p.bih[n], b2 = p.bhh[n];
#pragma unroll
            for (int r = 0; r < 8; ++r) {
                int row = hi * 8 + r;
                gi[row * 576 + n] = ai[r] + b1;
                gh[row * 576 + n] = ah[r] + b2;
            }
        }
        __syncthreads();
        // GRU gates -> tmpf
        for (int e = tid; e < 16 * NH; e += 256) {
            int row = e / NH, c = e - row * NH;
            float ir  = gi[row * 576 + c],          hr_ = gh[row * 576 + c];
            float iz  = gi[row * 576 + NH + c],     hz_ = gh[row * 576 + NH + c];
            float in_ = gi[row * 576 + 2 * NH + c], hn_ = gh[row * 576 + 2 * NH + c];
            float rr = sigm_f(ir + hr_);
            float zz = sigm_f(iz + hz_);
            float nn = tanhf(in_ + rr * hn_);
            tmpf[e] = (1.0f - zz) * nn + zz * hrf[e];
        }
        __syncthreads();
        // layernorm: one wave per row (2 rows per wave)
        for (int rr = 0; rr < 2; ++rr) {
            const int base = (wave + rr * 8) * NH;
            float s = 0.0f, s2 = 0.0f;
            for (int i = lane; i < NH; i += 32) {
                float v = tmpf[base + i];
                s += v; s2 += v * v;
            }
            s = wsum(s); s2 = wsum(s2);
            float mean = s * (1.0f / NH);
            float inv = rsqrtf(s2 * (1.0f / NH) - mean * mean + 1e-5f);
            for (int i = lane; i < NH; i += 32)
                hrf[base + i] = (tmpf[base + i] - mean) * inv * p.rlg[i] + p.rlb[i];
        }
        __syncthreads();
        // rho/phi heads: one wave per row, both dots in one pass
        for (int rr = 0; rr < 2; ++rr) {
            const int row = wave + rr * 8;
            float s0 = 0.0f, s1 = 0.0f;
            for (int i = lane; i < NH; i += 32) {
                float h = hrf[row * NH + i];
                s0 += h * p.fqw[i];
                s1 += h * p.fqw[NH + i];
            }
            s0 = wsum(s0); s1 = wsum(s1);
            if (lane == 0) {
                rho2[row] = sigm_f(s0 + p.fqb[0]) * 1.25f;
                phi2[row] = PI_F * tanhf(s1 + p.fqb[1]);
            }
        }
        __syncthreads();
        // rotate curr and emit preds[b, step, :]
        for (int e = tid; e < 16 * 32; e += 256) {
            int row = e >> 5, d = e & 31;
            float r = rho2[row], ph = phi2[row];
            float cc = cosf(ph), ss = sinf(ph);
            float re = cur[row * ND + d], im = cur[row * ND + 32 + d];
            float nre = r * (cc * re - ss * im);
            float nim = r * (ss * re + cc * im);
            cur[row * ND + d]      = nre;
            cur[row * ND + 32 + d] = nim;
            size_t ob = ((size_t)(b0 + row) * NWOUT + step) * ND;
            p.out[ob + d]      = nre;
            p.out[ob + 32 + d] = nim;
        }
        __syncthreads();
    }
}

extern "C" void kernel_launch(void* const* d_in, const int* in_sizes, int n_in,
                              void* d_out, int out_size, void* d_ws, size_t ws_size,
                              hipStream_t stream)
{
    (void)n_in; (void)out_size; (void)ws_size;
    const int nb = in_sizes[0] / (NQ * ND);       // 4096

    float* xpost_g = (float*)d_ws;                      // nb*64  f32
    float* hlast_g = xpost_g + (size_t)nb * ND;         // nb*192 f32
    __bf16* wb     = (__bf16*)(hlast_g + (size_t)nb * NH);  // bf16 weight arena

    P0 p0;
    p0.ciw   = (const float*)d_in[2];
    p0.b0c1w = (const float*)d_in[4];  p0.b0c2w = (const float*)d_in[6];
    p0.b1c1w = (const float*)d_in[10]; p0.b1c2w = (const float*)d_in[12];
    p0.fgw   = (const float*)d_in[18];
    p0.riw   = (const float*)d_in[22];
    p0.wih   = (const float*)d_in[24]; p0.whh   = (const float*)d_in[25];
    p0.wb    = wb;

    P1 p1;
    p1.x     = (const float*)d_in[0];
    p1.cib   = (const float*)d_in[3];
    p1.b0c1b = (const float*)d_in[5];  p1.b0c2b = (const float*)d_in[7];
    p1.b0g   = (const float*)d_in[8];  p1.b0b   = (const float*)d_in[9];
    p1.b1c1b = (const float*)d_in[11]; p1.b1c2b = (const float*)d_in[13];
    p1.b1g   = (const float*)d_in[14]; p1.b1b   = (const float*)d_in[15];
    p1.olg   = (const float*)d_in[16]; p1.olb   = (const float*)d_in[17];
    p1.fgb   = (const float*)d_in[19];
    p1.frw   = (const float*)d_in[20]; p1.frb   = (const float*)d_in[21];
    p1.wb    = wb;
    p1.xpost_g = xpost_g; p1.hlast_g = hlast_g;

    P2 p2;
    p2.rib = (const float*)d_in[23];
    p2.bih = (const float*)d_in[26]; p2.bhh = (const float*)d_in[27];
    p2.rlg = (const float*)d_in[28]; p2.rlb = (const float*)d_in[29];
    p2.fqw = (const float*)d_in[30]; p2.fqb = (const float*)d_in[31];
    p2.wb  = wb;
    p2.xpost_g = xpost_g; p2.hlast_g = hlast_g;
    p2.out = (float*)d_out;

    k0_prep<<<dim3((442368 + 255) / 256), dim3(256), 0, stream>>>(p0);
    k1_front<<<dim3(nb), dim3(256), 0, stream>>>(p1);
    k2_roll<<<dim3(nb / 16), dim3(256), 0, stream>>>(p2);
}
